// Elman_MD_68521908240468
// MI455X (gfx1250) — compile-verified, matrix operand order checked by reading, not run on hardware
//
#include <hip/hip_runtime.h>
#include <math.h>

// Problem constants (from reference)
#define TSTEPS 200
#define BATCH  256
#define IN_SZ  4
#define HID    1000
#define OUT_N  2
#define HP     1024   // padded hidden dim (multiple of 32 for WMMA K/N tiling)

// LDS weight slab: 64 N-rows x 1024 K halfs, padded +4 dwords every 256 dwords
// -> row stride 520 dwords (1040 halfs): banks spread 8 apart, 16B chunks stay aligned.
#define LDS_ROW_H 1040
#define LDS_ELEMS (64 * LDS_ROW_H)   // 133,120 B of the 320 KB WGP LDS

typedef __attribute__((ext_vector_type(16))) __bf16         v16bf;
typedef __attribute__((ext_vector_type(8)))  __bf16         v8bf;
typedef __attribute__((ext_vector_type(8)))  float          v8f;
typedef __attribute__((ext_vector_type(8)))  unsigned short v8us;
typedef __attribute__((ext_vector_type(4)))  unsigned int   u32x4;
typedef __attribute__((ext_vector_type(8)))  int            i32x8;
typedef __attribute__((ext_vector_type(4)))  int            i32x4;

#if defined(__has_builtin)
# if __has_builtin(__builtin_amdgcn_tensor_load_to_lds)
#  define HAVE_TDM 1
# endif
#endif
#ifndef HAVE_TDM
# define HAVE_TDM 0
#endif

__device__ __forceinline__ unsigned short f32_bf16(float x) {
    union { float f; unsigned int u; } a; a.f = x;
    unsigned int u = a.u + 0x7FFFu + ((a.u >> 16) & 1u);   // RNE
    return (unsigned short)(u >> 16);
}
__device__ __forceinline__ float bf16_f32(unsigned short h) {
    union { unsigned int u; float f; } a; a.u = ((unsigned int)h) << 16;
    return a.f;
}

// element index into padded LDS slab (r = local N row, k = K half-index)
__device__ __forceinline__ int lds_idx(int r, int k) {
    return r * LDS_ROW_H + k + ((k >= 512) ? 8 : 0);   // +4dw pad after 256dw
}

// Load one 16x32 bf16 WMMA A/B fragment from a row-major (ld = HP) bf16 buffer.
// CDNA5 16-bit operand layout: lanes 0-15 hold rows 0-15 with K {0..7, 16..23},
// lanes 16-31 hold rows 0-15 with K {8..15, 24..31}  -> two 16B loads per lane.
__device__ __forceinline__ v16bf load_frag_g(const unsigned short* __restrict__ base,
                                             int row0, int k0, int lane) {
    const size_t r  = (size_t)(row0 + (lane & 15));
    const int    kb = k0 + ((lane >> 4) << 3);
    const v8bf* q = reinterpret_cast<const v8bf*>(base + r * HP + kb);
    v8bf lo = q[0];          // K kb .. kb+7
    v8bf hi = q[2];          // K kb+16 .. kb+23
    v16bf f;
#pragma unroll
    for (int i = 0; i < 8; ++i) { f[i] = lo[i]; f[i + 8] = hi[i]; }
    return f;
}

// --- pack W_hh (HIDxHID f32, row n, col k) -> bf16 padded HPxHP, zeros outside ---
__global__ void k_pack_whh(const float* __restrict__ W, unsigned short* __restrict__ Wbf) {
    int idx = blockIdx.x * blockDim.x + threadIdx.x;   // 0 .. HP*HP-1
    int n = idx >> 10, k = idx & (HP - 1);
    float v = (n < HID && k < HID) ? W[n * HID + k] : 0.0f;
    Wbf[idx] = f32_bf16(v);
}

__global__ void k_zero_us(unsigned short* __restrict__ p) {
    p[blockIdx.x * blockDim.x + threadIdx.x] = 0;
}

// --- one recurrence step: h_new = tanh(x_t@W_in^T + b_in + b_hh + h_old@W_hh^T) ---
// Block: 256 thr = 8 waves as 4(M) x 2(N); wave tile 16x32; block tile 64x64.
// Grid: (BATCH/64, HP/64) = (4, 16).  Whole 64x1024 weight slab staged in LDS once
// (TDM async-tensor DMA), then the full K loop runs with ZERO further barriers.
__global__ __launch_bounds__(256)
void k_step(const unsigned short* __restrict__ hsrc,  // (BATCH, HP) bf16
            unsigned short* __restrict__       hdst,  // (BATCH, HP) bf16
            const unsigned short* __restrict__ Wbf,   // (HP, HP) bf16, n-major
            const float* __restrict__ input,          // (T, BATCH, 4) f32
            const float* __restrict__ W_in,           // (HID, 4) f32
            const float* __restrict__ b_in,           // (HID,)
            const float* __restrict__ b_hh,           // (HID,)
            int t)
{
    __shared__ __align__(16) unsigned short Bs[LDS_ELEMS];

    const int tid  = threadIdx.x;
    const int lane = tid & 31;
    const int wid  = tid >> 5;                            // 0..7
    const int m0   = blockIdx.x * 64 + (wid >> 1) * 16;   // batch-row base of wave
    const int nblk = blockIdx.y * 64;                     // N base of block
    const int wn   = wid & 1;                             // N-wave (0/1)
    const int n0   = nblk + wn * 32;

    // ---- stage the block's full weight slab (64 rows x 1024 K) into LDS ----
#if HAVE_TDM
    if (wid == 0) {
        const unsigned long long ga =
            (unsigned long long)(uintptr_t)(Wbf + (size_t)nblk * HP);
        const unsigned lds_base = (unsigned)(uintptr_t)&Bs[0];  // low 32b = LDS offset
        // D# group 0: count=1 | lds_addr | global_addr | type=2
        u32x4 g0;
        g0[0] = 1u;
        g0[1] = lds_base;
        g0[2] = (unsigned)ga;
        g0[3] = (unsigned)((ga >> 32) & 0x1FFFFFFu) | (2u << 30);
        // D# group 1: data_size=2B, pad 4dw every 256dw, dims/strides in elements
        i32x8 g1;
        g1[0] = (int)((1u << 16) | (1u << 20) | (7u << 22) | (3u << 25));
        g1[1] = (int)(1024u << 16);   // tensor_dim0 = 1024 (lo16)
        g1[2] = (int)(1024u << 16);   // tensor_dim1 = 1024 (lo16)
        g1[3] = (int)(1024u << 16);   // tile_dim0 = 1024
        g1[4] = 64;                   // tile_dim1 = 64, tile_dim2 = 0
        g1[5] = 1024;                 // tensor_dim0_stride = 1024
        g1[6] = 0;
        g1[7] = 0;
        i32x4 gz = {0, 0, 0, 0};      // groups 2/3 unused (2D tensor)
#if __clang_major__ >= 23
        i32x8 gz8 = {0, 0, 0, 0, 0, 0, 0, 0};
        __builtin_amdgcn_tensor_load_to_lds(g0, g1, gz, gz, gz8, 0);
#else
        __builtin_amdgcn_tensor_load_to_lds(g0, g1, gz, gz, 0);
#endif
        __builtin_amdgcn_s_wait_tensorcnt(0);
    }
#else
    // cooperative fallback: 8192 16B chunks, identical padded layout
    for (int c = tid; c < 64 * 128; c += 256) {
        int row = c >> 7, k = (c & 127) << 3;
        *reinterpret_cast<v8us*>(&Bs[lds_idx(row, k)]) =
            *reinterpret_cast<const v8us*>(&Wbf[(size_t)(nblk + row) * HP + k]);
    }
#endif
    __syncthreads();   // the only barrier in the whole step

    // ---- K loop: A from L2-resident h_old, B from LDS, 2 WMMAs per chunk ----
    v8f acc0 = {}, acc1 = {};
    const int kb_half = (lane >> 4) << 3;   // 0 or 8 (lane-half K base)
    for (int k0 = 0; k0 < HP; k0 += 32) {
        if (k0 + 64 < HP)
            __builtin_prefetch(&hsrc[(size_t)(m0 + (lane & 15)) * HP + k0 + 64], 0, 1);
        v16bf a = load_frag_g(hsrc, m0, k0, lane);
        const int kb = k0 + kb_half;
#pragma unroll
        for (int tj = 0; tj < 2; ++tj) {
            const int r = wn * 32 + tj * 16 + (lane & 15);   // local N row
            v8bf lo = *reinterpret_cast<const v8bf*>(&Bs[lds_idx(r, kb)]);
            v8bf hi = *reinterpret_cast<const v8bf*>(&Bs[lds_idx(r, kb + 16)]);
            v16bf b;
#pragma unroll
            for (int i = 0; i < 8; ++i) { b[i] = lo[i]; b[i + 8] = hi[i]; }
            if (tj == 0)
                acc0 = __builtin_amdgcn_wmma_f32_16x16x32_bf16(false, a, false, b,
                                                               (short)0, acc0, false, false);
            else
                acc1 = __builtin_amdgcn_wmma_f32_16x16x32_bf16(false, a, false, b,
                                                               (short)0, acc1, false, false);
        }
    }

    // ---- epilogue: + input projection + biases, tanh, store bf16 (mask n<HID) ----
    // C/D layout: lane L, vgpr v -> row m0 + ((L>=16)?8:0) + v, col (L&15).
    const float* xin = input + (size_t)t * BATCH * IN_SZ;
    const int rbase = m0 + ((lane >> 4) << 3);
#pragma unroll
    for (int tj = 0; tj < 2; ++tj) {
        const v8f& acc = tj ? acc1 : acc0;
        const int col = n0 + tj * 16 + (lane & 15);
        const bool ok = (col < HID);
        float4 wrow = make_float4(0.f, 0.f, 0.f, 0.f);
        float  bias = 0.f;
        if (ok) {
            wrow = reinterpret_cast<const float4*>(W_in)[col];   // W_in row = 4 floats
            bias = b_in[col] + b_hh[col];
        }
#pragma unroll
        for (int v = 0; v < 8; ++v) {
            const int b = rbase + v;
            float4 x = reinterpret_cast<const float4*>(xin)[b];
            float xp = bias + x.x * wrow.x + x.y * wrow.y + x.z * wrow.z + x.w * wrow.w;
            float hv = tanhf(acc[v] + xp);
            if (ok) hdst[(size_t)b * HP + col] = f32_bf16(hv);
        }
    }
}

// --- FC head for one timestep: out[t,b,o] = tanh(b_fc[o] + sum_k h[b,k]*W_fc[o,k]) ---
// One wave per (b,o) pair: 512 waves = 64 blocks x 8 waves, wave32 shfl reduction.
__global__ __launch_bounds__(256)
void k_out(const unsigned short* __restrict__ h,     // (BATCH, HP) bf16
           const float* __restrict__ W_fc,           // (OUT_N, HID)
           const float* __restrict__ b_fc,           // (OUT_N,)
           float* __restrict__ out, int t)
{
    const int lane = threadIdx.x & 31;
    const int w = blockIdx.x * 8 + (threadIdx.x >> 5);  // 0..511
    const int b = w >> 1, o = w & 1;
    float s = 0.f;
    for (int k = lane; k < HID; k += 32)
        s += bf16_f32(h[(size_t)b * HP + k]) * W_fc[o * HID + k];
    s += __shfl_xor(s, 16);
    s += __shfl_xor(s, 8);
    s += __shfl_xor(s, 4);
    s += __shfl_xor(s, 2);
    s += __shfl_xor(s, 1);
    if (lane == 0)
        out[((size_t)t * BATCH + b) * OUT_N + o] = tanhf(s + b_fc[o]);
}

extern "C" void kernel_launch(void* const* d_in, const int* in_sizes, int n_in,
                              void* d_out, int out_size, void* d_ws, size_t ws_size,
                              hipStream_t stream)
{
    (void)in_sizes; (void)n_in; (void)out_size; (void)ws_size;
    const float* input = (const float*)d_in[0];
    /* d_in[1] = target (unused in forward) */
    const float* W_in  = (const float*)d_in[2];
    const float* b_in  = (const float*)d_in[3];
    const float* W_hh  = (const float*)d_in[4];
    const float* b_hh  = (const float*)d_in[5];
    const float* W_fc  = (const float*)d_in[6];
    const float* b_fc  = (const float*)d_in[7];
    float* out = (float*)d_out;

    // Workspace layout (bf16 as ushort): Wbf[HP*HP] | h0[BATCH*HP] | h1[BATCH*HP]  (~3 MB)
    unsigned short* Wbf = (unsigned short*)d_ws;
    unsigned short* h0  = Wbf + (size_t)HP * HP;
    unsigned short* h1  = h0 + (size_t)BATCH * HP;

    k_pack_whh<<<(HP * HP) / 256, 256, 0, stream>>>(W_hh, Wbf);
    k_zero_us<<<(2 * BATCH * HP) / 256, 256, 0, stream>>>(h0);  // zeros h0 and h1

    for (int t = 0; t < TSTEPS; ++t) {
        const unsigned short* hs = (t & 1) ? h1 : h0;
        unsigned short*       hd = (t & 1) ? h0 : h1;
        k_step<<<dim3(BATCH / 64, HP / 64), 256, 0, stream>>>(
            hs, hd, Wbf, input, W_in, b_in, b_hh, t);
        k_out<<<64, 256, 0, stream>>>(hd, W_fc, b_fc, out, t);
    }
}
